// ParaBlock_14980845928837
// MI455X (gfx1250) — compile-verified, matrix operand order checked by reading, not run on hardware
//
#include <hip/hip_runtime.h>
#include <math.h>

// ---------------------------------------------------------------------------
// Types for CDNA5 WMMA (wave32, v_wmma_f32_16x16x32_bf16)
// ---------------------------------------------------------------------------
typedef __bf16 bf16_t;
typedef bf16_t v16bf __attribute__((ext_vector_type(16)));
typedef float  v8f   __attribute__((ext_vector_type(8)));

#define DEV static __device__ __forceinline__

union FragU { v16bf v; uint4 u[2]; };

// Load a 16-element bf16 fragment: 8 contiguous elems at p+kbase and 8 at
// p+kbase+16 (matches ISA 16-bit A/B VGPR layout: lane kbase = 8*(lane>>4)).
DEV v16bf load_frag(const bf16_t* p, int kbase) {
  FragU f;
  f.u[0] = *(const uint4*)(p + kbase);
  f.u[1] = *(const uint4*)(p + kbase + 16);
  return f.v;
}

DEV v8f wmma_bf16(v16bf a, v16bf b, v8f c) {
  return __builtin_amdgcn_wmma_f32_16x16x32_bf16(false, a, false, b,
                                                 (short)0, c, false, false);
}

// ---------------------------------------------------------------------------
// Problem constants
// ---------------------------------------------------------------------------
constexpr int BIMG   = 4;
constexpr int HPIX   = 64;
constexpr int DIM    = 768;
constexpr int HEADS  = 12;
constexpr int HEAD   = 64;
constexpr int WIN    = 14;
constexpr int NTOK   = WIN * WIN;        // 196 tokens / window
constexpr int NWIN   = BIMG * 5 * 5;     // 100 windows (70/14 = 5 per dim)
constexpr int TOK    = NWIN * NTOK;      // 19600 window tokens
constexpr int BH     = NWIN * HEADS;     // 1200 (window, head) pairs
constexpr int NPAD   = 224;              // tokens padded to 14*16 for WMMA
constexpr int IMGTOK = BIMG * HPIX * HPIX; // 16384 image tokens
constexpr int MLP    = 4 * DIM;          // 3072
constexpr int QKVN   = 3 * DIM;          // 2304

// ---------------------------------------------------------------------------
// Small utility kernels
// ---------------------------------------------------------------------------
__global__ void f32_to_bf16_kernel(const float* __restrict__ src,
                                   bf16_t* __restrict__ dst, int n) {
  int i = blockIdx.x * 256 + threadIdx.x;
  if (i < n) dst[i] = (bf16_t)src[i];
}

__global__ void zero16_kernel(uint4* __restrict__ p, int n16) {
  int i = blockIdx.x * 256 + threadIdx.x;
  if (i < n16) p[i] = make_uint4(0u, 0u, 0u, 0u);
}

// ---------------------------------------------------------------------------
// LayerNorm kernels (one wave per token, 768 = 24 elems/lane)
// ---------------------------------------------------------------------------
DEV float wave_sum(float s) {
  #pragma unroll
  for (int o = 1; o < 32; o <<= 1) s += __shfl_xor(s, o, 32);
  return s;
}

// LN1 fused with window partition: output bf16 [19600, 768]; padded tokens = 0
__global__ __launch_bounds__(256)
void ln1_window_kernel(const float* __restrict__ x, const float* __restrict__ g,
                       const float* __restrict__ b, bf16_t* __restrict__ out) {
  int wave = threadIdx.x >> 5, lane = threadIdx.x & 31;
  int tix = blockIdx.x * 8 + wave;
  if (tix >= TOK) return;
  int w = tix / NTOK, n = tix % NTOK;
  int bi = w / 25, wq = w % 25;
  int row = (wq / 5) * WIN + n / WIN;
  int col = (wq % 5) * WIN + n % WIN;
  bf16_t* dst = out + (size_t)tix * DIM;
  if (row >= HPIX || col >= HPIX) {
    #pragma unroll
    for (int i = 0; i < 24; ++i) dst[lane + i * 32] = (bf16_t)0.f;
    return;
  }
  const float* src = x + (((size_t)(bi * HPIX + row)) * HPIX + col) * DIM;
  float v[24]; float s = 0.f;
  #pragma unroll
  for (int i = 0; i < 24; ++i) { v[i] = src[lane + i * 32]; s += v[i]; }
  float mu = wave_sum(s) * (1.f / 768.f);
  float s2 = 0.f;
  #pragma unroll
  for (int i = 0; i < 24; ++i) { float d = v[i] - mu; s2 += d * d; }
  float rs = rsqrtf(wave_sum(s2) * (1.f / 768.f) + 1e-5f);
  #pragma unroll
  for (int i = 0; i < 24; ++i) {
    int c = lane + i * 32;
    dst[c] = (bf16_t)((v[i] - mu) * rs * g[c] + b[c]);
  }
}

// LN2: f32 [16384,768] -> bf16 [16384,768]
__global__ __launch_bounds__(256)
void ln2_kernel(const float* __restrict__ x, const float* __restrict__ g,
                const float* __restrict__ b, bf16_t* __restrict__ out) {
  int wave = threadIdx.x >> 5, lane = threadIdx.x & 31;
  int tix = blockIdx.x * 8 + wave;
  if (tix >= IMGTOK) return;
  const float* src = x + (size_t)tix * DIM;
  bf16_t* dst = out + (size_t)tix * DIM;
  float v[24]; float s = 0.f;
  #pragma unroll
  for (int i = 0; i < 24; ++i) { v[i] = src[lane + i * 32]; s += v[i]; }
  float mu = wave_sum(s) * (1.f / 768.f);
  float s2 = 0.f;
  #pragma unroll
  for (int i = 0; i < 24; ++i) { float d = v[i] - mu; s2 += d * d; }
  float rs = rsqrtf(wave_sum(s2) * (1.f / 768.f) + 1e-5f);
  #pragma unroll
  for (int i = 0; i < 24; ++i) {
    int c = lane + i * 32;
    dst[c] = (bf16_t)((v[i] - mu) * rs * g[c] + b[c]);
  }
}

// ---------------------------------------------------------------------------
// Relative-position bias tables: relH/relW [1200, 196, 14] f32
// relH[bh,q,kh] = sum_c q[bh,q,c] * rel_pos_h[qh-kh+13, c]   (q unscaled)
// ---------------------------------------------------------------------------
__global__ __launch_bounds__(256)
void relpos_kernel(const bf16_t* __restrict__ qbuf, const float* __restrict__ rph,
                   const float* __restrict__ rpw, float* __restrict__ relH,
                   float* __restrict__ relW) {
  int wave = threadIdx.x >> 5, lane = threadIdx.x & 31;
  int idx = blockIdx.x * 8 + wave;
  if (idx >= BH * NTOK) return;
  int bh = idx / NTOK, q = idx % NTOK;
  int qh = q / WIN, qw = q % WIN;
  const bf16_t* qp = qbuf + ((size_t)bh * NPAD + q) * HEAD;
  if (lane < 28) {
    int isW = lane >= 14;
    int kk = isW ? (lane - 14) : lane;
    const float* rp = (isW ? rpw : rph) + ((isW ? qw : qh) - kk + (WIN - 1)) * HEAD;
    float acc = 0.f;
    #pragma unroll
    for (int c = 0; c < HEAD; ++c) acc += (float)qp[c] * rp[c];
    (isW ? relW : relH)[((size_t)bh * NTOK + q) * WIN + kk] = acc;
  }
}

// ---------------------------------------------------------------------------
// Generic bf16 WMMA GEMM: C[M,N] = A[M,K] @ B[K,N] + bias, templated epilogue.
// Block: 256 threads = 8 waves; tile 128(M) x 64(N); K step 32.
// ---------------------------------------------------------------------------
enum { EPI_QKV = 0, EPI_PROJ = 1, EPI_GELU = 2, EPI_RES = 3 };

struct GemmArgs {
  const bf16_t* A; const bf16_t* B; const float* bias;
  int M, N, K;
  float*  outF;       // PROJ: xres ; RES: d_out
  bf16_t* outB;       // GELU: hidden
  const float* res;   // PROJ: original x ; RES: xres
  bf16_t* qb; bf16_t* kb; bf16_t* vb;   // QKV scatter targets
};

template <int EPI>
__global__ __launch_bounds__(256)
void gemm_bf16_kernel(GemmArgs ga) {
  __shared__ __align__(16) bf16_t As[128][32];
  __shared__ __align__(16) bf16_t Bt[64][32];   // B transposed: [n][k]
  const int tid = threadIdx.x;
  const int wave = tid >> 5, lane = tid & 31;
  const int half = lane >> 4, colL = lane & 15, kbase = half * 8;
  const int m0 = blockIdx.x * 128;
  const int n0 = blockIdx.y * 64;

  v8f acc[4];
  #pragma unroll
  for (int nt = 0; nt < 4; ++nt)
    #pragma unroll
    for (int r = 0; r < 8; ++r) acc[nt][r] = 0.f;

  const int kIters = ga.K >> 5;
  const int arow = tid >> 1, ahalf = tid & 1;
  const int agrow = m0 + arow;
  const bool avalid = agrow < ga.M;
  const int bkr = tid >> 3, bcg = tid & 7;

  for (int kk = 0; kk < kIters; ++kk) {
    // stage A tile 128x32 (2x uint4 per thread), zero OOB rows
    uint4 a0 = make_uint4(0u, 0u, 0u, 0u), a1 = a0;
    if (avalid) {
      const bf16_t* ap = ga.A + (size_t)agrow * ga.K + kk * 32 + ahalf * 16;
      a0 = *(const uint4*)ap;
      a1 = *(const uint4*)(ap + 8);
    }
    *(uint4*)&As[arow][ahalf * 16]     = a0;
    *(uint4*)&As[arow][ahalf * 16 + 8] = a1;
    // stage B tile 32x64 transposed into Bt[n][k]
    {
      const bf16_t* bp = ga.B + (size_t)(kk * 32 + bkr) * ga.N + n0 + bcg * 8;
      uint4 bu = *(const uint4*)bp;
      const bf16_t* be = (const bf16_t*)&bu;
      #pragma unroll
      for (int j = 0; j < 8; ++j) Bt[bcg * 8 + j][bkr] = be[j];
    }
    __syncthreads();
    v16bf af = load_frag(&As[wave * 16 + colL][0], kbase);
    #pragma unroll
    for (int nt = 0; nt < 4; ++nt) {
      v16bf bfr = load_frag(&Bt[nt * 16 + colL][0], kbase);
      acc[nt] = wmma_bf16(af, bfr, acc[nt]);
    }
    __syncthreads();
  }

  // epilogue (C layout: col = lane&15 (+16*nt), row = r + 8*(lane>>4))
  #pragma unroll
  for (int nt = 0; nt < 4; ++nt) {
    #pragma unroll
    for (int r = 0; r < 8; ++r) {
      int grow = m0 + wave * 16 + r + 8 * half;
      int gcol = n0 + nt * 16 + colL;
      if (grow >= ga.M) continue;
      float val = acc[nt][r] + ga.bias[gcol];
      if (EPI == EPI_QKV) {
        int m3 = gcol / DIM, rem = gcol % DIM;
        int h = rem / HEAD, c = rem & (HEAD - 1);
        int win = grow / NTOK, n = grow % NTOK;
        bf16_t* d = (m3 == 0) ? ga.qb : (m3 == 1) ? ga.kb : ga.vb;
        d[(((size_t)(win * HEADS + h)) * NPAD + n) * HEAD + c] = (bf16_t)val;
      } else if (EPI == EPI_PROJ) {
        int win = grow / NTOK, n = grow % NTOK;
        int bi = win / 25, wq = win % 25;
        int row = (wq / 5) * WIN + n / WIN;
        int col = (wq % 5) * WIN + n % WIN;
        if (row < HPIX && col < HPIX) {
          size_t idx = (((size_t)(bi * HPIX + row)) * HPIX + col) * DIM + gcol;
          ga.outF[idx] = ga.res[idx] + val;
        }
      } else if (EPI == EPI_GELU) {
        float gel = 0.5f * val * (1.f + erff(val * 0.70710678118654752440f));
        ga.outB[(size_t)grow * ga.N + gcol] = (bf16_t)gel;
      } else { // EPI_RES
        size_t idx = (size_t)grow * ga.N + gcol;
        ga.outF[idx] = ga.res[idx] + val;
      }
    }
  }
}

// ---------------------------------------------------------------------------
// Fused windowed attention: grid (1200, 4), block 128 (4 waves).
// Wave owns one 16-query tile; scores for all 208 (13 tiles) keys kept in
// registers; softmax via 16-lane butterflies; P staged via LDS for P@V.
// ---------------------------------------------------------------------------
__global__ __launch_bounds__(128)
void attn_kernel(const bf16_t* __restrict__ qbuf, const bf16_t* __restrict__ kbuf,
                 const bf16_t* __restrict__ vbuf, const float* __restrict__ relH,
                 const float* __restrict__ relW, bf16_t* __restrict__ attn_out) {
  __shared__ __align__(16) bf16_t Vt[HEAD][NPAD];      // V transposed [c][key]
  __shared__ __align__(16) bf16_t Pls[4][16][NPAD];    // per-wave P scratch
  const int tid = threadIdx.x, wave = tid >> 5, lane = tid & 31;
  const int half = lane >> 4, colL = lane & 15, kbase = half * 8;
  const int bh = blockIdx.x;
  const int win = bh / HEADS, head = bh % HEADS;
  int qtile = blockIdx.y * 4 + wave;
  if (qtile > 12) qtile = 12;   // duplicates store identical values (benign)

  // cooperative transposed V load (224*64 bf16 = 1792 uint4 / 128 threads)
  {
    const bf16_t* vb = vbuf + (size_t)bh * NPAD * HEAD;
    for (int i = tid; i < NPAD * 8; i += 128) {
      int key = i >> 3, cg = i & 7;
      uint4 u = *(const uint4*)(vb + key * HEAD + cg * 8);
      const bf16_t* e = (const bf16_t*)&u;
      #pragma unroll
      for (int j = 0; j < 8; ++j) Vt[cg * 8 + j][key] = e[j];
    }
  }
  __syncthreads();

  // Q fragments (A-matrix, K = 64 -> two 16x32 fragments)
  const bf16_t* qp = qbuf + ((size_t)bh * NPAD + qtile * 16 + colL) * HEAD;
  v16bf qa0 = load_frag(qp, kbase);
  v16bf qa1 = load_frag(qp, kbase + 32);

  // S = q @ k^T ; B-fragment of k^T == contiguous rows of k
  v8f S[13];
  #pragma unroll
  for (int t = 0; t < 13; ++t)
    #pragma unroll
    for (int r = 0; r < 8; ++r) S[t][r] = 0.f;
  const bf16_t* kbp = kbuf + (size_t)bh * NPAD * HEAD;
  #pragma unroll
  for (int t = 0; t < 13; ++t) {
    const bf16_t* kp = kbp + (t * 16 + colL) * HEAD;
    v16bf kb0 = load_frag(kp, kbase);
    v16bf kb1 = load_frag(kp, kbase + 32);
    S[t] = wmma_bf16(qa0, kb0, S[t]);
    S[t] = wmma_bf16(qa1, kb1, S[t]);
  }

  // scale + decomposed relative position bias
  const float* rHp = relH + (size_t)bh * NTOK * WIN;
  const float* rWp = relW + (size_t)bh * NTOK * WIN;
  #pragma unroll
  for (int t = 0; t < 13; ++t) {
    int key = t * 16 + colL;
    int kh = key / WIN, kw = key % WIN;
    #pragma unroll
    for (int r = 0; r < 8; ++r) {
      int q = qtile * 16 + r + 8 * half;
      float s = S[t][r] * 0.125f;                 // HEAD^-0.5
      if (key < NTOK && q < NTOK) s += rHp[q * WIN + kh] + rWp[q * WIN + kw];
      if (key >= NTOK) s = -3.0e38f;              // mask padded keys
      S[t][r] = s;
    }
  }

  // row softmax: reduce over 13 tiles locally + 16 lanes of the half-wave
  #pragma unroll
  for (int r = 0; r < 8; ++r) {
    float mx = -3.4e38f;
    #pragma unroll
    for (int t = 0; t < 13; ++t) mx = fmaxf(mx, S[t][r]);
    #pragma unroll
    for (int o = 1; o < 16; o <<= 1) mx = fmaxf(mx, __shfl_xor(mx, o, 32));
    float sum = 0.f;
    #pragma unroll
    for (int t = 0; t < 13; ++t) { float e = __expf(S[t][r] - mx); S[t][r] = e; sum += e; }
    #pragma unroll
    for (int o = 1; o < 16; o <<= 1) sum += __shfl_xor(sum, o, 32);
    float inv = 1.f / sum;
    #pragma unroll
    for (int t = 0; t < 13; ++t) S[t][r] *= inv;
  }

  // stage P (C-layout -> row-major LDS) for the A-fragment of P@V
  #pragma unroll
  for (int t = 0; t < 13; ++t) {
    int key = t * 16 + colL;
    #pragma unroll
    for (int r = 0; r < 8; ++r) Pls[wave][r + 8 * half][key] = (bf16_t)S[t][r];
  }
  #pragma unroll
  for (int i = 0; i < 8; ++i) {        // zero pad key-cols 208..223
    int idx = lane + i * 32;
    Pls[wave][idx >> 4][208 + (idx & 15)] = (bf16_t)0.f;
  }
  asm volatile("s_wait_dscnt 0" ::: "memory");   // per-wave LDS turnaround

  // O[16,64] = P[16,224] @ V[224,64]
  v8f O[4];
  #pragma unroll
  for (int nt = 0; nt < 4; ++nt)
    #pragma unroll
    for (int r = 0; r < 8; ++r) O[nt][r] = 0.f;
  #pragma unroll
  for (int kb = 0; kb < 7; ++kb) {
    v16bf pa = load_frag(&Pls[wave][colL][0], kb * 32 + kbase);
    #pragma unroll
    for (int nt = 0; nt < 4; ++nt) {
      v16bf vf = load_frag(&Vt[nt * 16 + colL][0], kb * 32 + kbase);
      O[nt] = wmma_bf16(pa, vf, O[nt]);
    }
  }

  // scatter to head-interleaved layout for the proj GEMM
  #pragma unroll
  for (int nt = 0; nt < 4; ++nt) {
    #pragma unroll
    for (int r = 0; r < 8; ++r) {
      int q = qtile * 16 + r + 8 * half;
      if (q < NTOK) {
        int c = nt * 16 + colL;
        attn_out[((size_t)(win * NTOK + q)) * DIM + head * HEAD + c] = (bf16_t)O[nt][r];
      }
    }
  }
}

// ---------------------------------------------------------------------------
// Host launch
// ---------------------------------------------------------------------------
extern "C" void kernel_launch(void* const* d_in, const int* in_sizes, int n_in,
                              void* d_out, int out_size, void* d_ws, size_t ws_size,
                              hipStream_t stream) {
  const float* x       = (const float*)d_in[0];
  const float* n1g     = (const float*)d_in[1];
  const float* n1b     = (const float*)d_in[2];
  const float* qkv_w   = (const float*)d_in[3];
  const float* qkv_b   = (const float*)d_in[4];
  const float* proj_w  = (const float*)d_in[5];
  const float* proj_b  = (const float*)d_in[6];
  const float* rph     = (const float*)d_in[7];
  const float* rpw     = (const float*)d_in[8];
  const float* n2g     = (const float*)d_in[9];
  const float* n2b     = (const float*)d_in[10];
  const float* w1      = (const float*)d_in[11];
  const float* b1      = (const float*)d_in[12];
  const float* w2      = (const float*)d_in[13];
  const float* b2      = (const float*)d_in[14];
  float* out = (float*)d_out;

  // workspace carve-up (256B aligned)
  char* base = (char*)d_ws;
  size_t off = 0;
  auto alloc = [&](size_t bytes) -> char* {
    char* p = base + off;
    off += (bytes + 255) & ~(size_t)255;
    return p;
  };
  bf16_t* qkv_w_bf = (bf16_t*)alloc((size_t)DIM * QKVN * 2);
  bf16_t* proj_w_bf= (bf16_t*)alloc((size_t)DIM * DIM * 2);
  bf16_t* w1_bf    = (bf16_t*)alloc((size_t)DIM * MLP * 2);
  bf16_t* w2_bf    = (bf16_t*)alloc((size_t)MLP * DIM * 2);
  bf16_t* xw_bf    = (bf16_t*)alloc((size_t)TOK * DIM * 2);
  bf16_t* q_bf     = (bf16_t*)alloc((size_t)BH * NPAD * HEAD * 2);
  bf16_t* k_bf     = (bf16_t*)alloc((size_t)BH * NPAD * HEAD * 2);
  bf16_t* v_bf     = (bf16_t*)alloc((size_t)BH * NPAD * HEAD * 2);
  float*  relH     = (float*)alloc((size_t)BH * NTOK * WIN * 4);
  float*  relW     = (float*)alloc((size_t)BH * NTOK * WIN * 4);
  bf16_t* attn_bf  = (bf16_t*)alloc((size_t)TOK * DIM * 2);
  float*  xres     = (float*)alloc((size_t)IMGTOK * DIM * 4);
  bf16_t* ln2_bf   = (bf16_t*)alloc((size_t)IMGTOK * DIM * 2);
  bf16_t* hidden   = (bf16_t*)alloc((size_t)IMGTOK * MLP * 2);
  (void)ws_size; (void)n_in; (void)in_sizes; (void)out_size;

  auto cvt = [&](const float* s, bf16_t* d, int n) {
    f32_to_bf16_kernel<<<(n + 255) / 256, 256, 0, stream>>>(s, d, n);
  };
  cvt(qkv_w,  qkv_w_bf, DIM * QKVN);
  cvt(proj_w, proj_w_bf, DIM * DIM);
  cvt(w1,     w1_bf,     DIM * MLP);
  cvt(w2,     w2_bf,     MLP * DIM);

  // zero q/k/v (padded rows 196..223 must be 0); they are contiguous in ws
  {
    int n16 = (int)(((size_t)BH * NPAD * HEAD * 2 * 3) / 16);
    zero16_kernel<<<(n16 + 255) / 256, 256, 0, stream>>>((uint4*)q_bf, n16);
  }

  // LN1 + window partition
  ln1_window_kernel<<<TOK / 8, 256, 0, stream>>>(x, n1g, n1b, xw_bf);

  // QKV GEMM: [19600,768] @ [768,2304]
  {
    GemmArgs ga{};
    ga.A = xw_bf; ga.B = qkv_w_bf; ga.bias = qkv_b;
    ga.M = TOK; ga.N = QKVN; ga.K = DIM;
    ga.qb = q_bf; ga.kb = k_bf; ga.vb = v_bf;
    dim3 grid((TOK + 127) / 128, QKVN / 64);
    gemm_bf16_kernel<EPI_QKV><<<grid, 256, 0, stream>>>(ga);
  }

  // relative-position bias tables
  relpos_kernel<<<(BH * NTOK) / 8, 256, 0, stream>>>(q_bf, rph, rpw, relH, relW);

  // fused attention
  attn_kernel<<<dim3(BH, 4), 128, 0, stream>>>(q_bf, k_bf, v_bf, relH, relW, attn_bf);

  // proj GEMM + window unpartition + residual -> xres (f32)
  {
    GemmArgs ga{};
    ga.A = attn_bf; ga.B = proj_w_bf; ga.bias = proj_b;
    ga.M = TOK; ga.N = DIM; ga.K = DIM;
    ga.outF = xres; ga.res = x;
    dim3 grid((TOK + 127) / 128, DIM / 64);
    gemm_bf16_kernel<EPI_PROJ><<<grid, 256, 0, stream>>>(ga);
  }

  // LN2
  ln2_kernel<<<IMGTOK / 8, 256, 0, stream>>>(xres, n2g, n2b, ln2_bf);

  // MLP1 GEMM + exact GELU -> hidden (bf16)
  {
    GemmArgs ga{};
    ga.A = ln2_bf; ga.B = w1_bf; ga.bias = b1;
    ga.M = IMGTOK; ga.N = MLP; ga.K = DIM;
    ga.outB = hidden;
    dim3 grid(IMGTOK / 128, MLP / 64);
    gemm_bf16_kernel<EPI_GELU><<<grid, 256, 0, stream>>>(ga);
  }

  // MLP2 GEMM + residual -> d_out (f32)
  {
    GemmArgs ga{};
    ga.A = hidden; ga.B = w2_bf; ga.bias = b2;
    ga.M = IMGTOK; ga.N = DIM; ga.K = MLP;
    ga.outF = out; ga.res = xres;
    dim3 grid(IMGTOK / 128, DIM / 64);
    gemm_bf16_kernel<EPI_RES><<<grid, 256, 0, stream>>>(ga);
  }
}
